// Decoder_12979391169054
// MI455X (gfx1250) — matrix operand order repaired and checked by reading, not compiled, not run on hardware
//
#include <hip/hip_runtime.h>
#include <math.h>

typedef _Float16 v16h __attribute__((ext_vector_type(16)));
typedef _Float16 v8h  __attribute__((ext_vector_type(8)));
typedef float    v8f  __attribute__((ext_vector_type(8)));
typedef int      v4i  __attribute__((vector_size(16)));

#define BATCH 16
#define LATN  512
#define NF    64
#define CHO   3
#define SS    128
#define HW    16384            /* SS*SS */
#define NC    8
#define FH    128
#define FPAR  37248
#define NSCALE 5
#define EMB   ((size_t)BATCH*NF*HW)   /* 16,777,216 floats per snapshot */
#define PI_2  1.57079632679489662f

/* packed swizzled f16 weight image per sample: W1|W2|W3|Ws */
#define WPK_W1 0
#define WPK_W2 8192
#define WPK_W3 24576
#define WPK_WS 32768
#define WPK_SZ 36864           /* f16 elements per sample (72 KB) */

__constant__ int d_ks[NSCALE] = {3,5,9,17,33};

/* offsets into per-sample hypernet param vector p[b][FPAR] */
#define OFF_W1 0
#define OFF_B1 8192
#define OFF_W2 8320
#define OFF_B2 24704
#define OFF_W3 24832
#define OFF_B3 33024
#define OFF_WS 33088
#define OFF_BS 37184

#if defined(__has_builtin)
# if __has_builtin(__builtin_amdgcn_global_load_async_to_lds_b128)
#  define USE_ASYNC_COPY 1
# endif
#endif
#ifndef USE_ASYNC_COPY
# define USE_ASYNC_COPY 0
#endif

/* ---------------- hypernet: p = lat @ dyna_w^T + dyna_b (once) ------------- */
__global__ void k_hyper(const float* __restrict__ lat, const float* __restrict__ dw,
                        const float* __restrict__ db, float* __restrict__ p) {
  __shared__ float slat[BATCH * LATN];
  for (int i = threadIdx.x; i < BATCH * LATN; i += blockDim.x) slat[i] = lat[i];
  __syncthreads();
  int j = blockIdx.x * blockDim.x + threadIdx.x;
  if (j >= FPAR) return;
  float acc[BATCH];
#pragma unroll
  for (int b = 0; b < BATCH; b++) acc[b] = 0.f;
  const float* wr = dw + (size_t)j * LATN;
  for (int k = 0; k < LATN; k++) {
    float w = wr[k];
#pragma unroll
    for (int b = 0; b < BATCH; b++) acc[b] += w * slat[b * LATN + k];
  }
  float bias = db[j];
  for (int b = 0; b < BATCH; b++) p[(size_t)b * FPAR + j] = acc[b] + bias;
}

/* ------------- L2-normalize fan-in of generated weight rows (once) --------- */
__global__ void k_wnorm(float* __restrict__ p) {
  int idx = blockIdx.x * blockDim.x + threadIdx.x;
  if (idx >= BATCH * 384) return;
  int b = idx / 384, r = idx % 384;
  int off, len, row;
  if (r < 128)      { off = OFF_W1; len = NF; row = r; }
  else if (r < 256) { off = OFF_W2; len = FH; row = r - 128; }
  else if (r < 320) { off = OFF_W3; len = FH; row = r - 256; }
  else              { off = OFF_WS; len = NF; row = r - 320; }
  float* w = p + (size_t)b * FPAR + off + row * len;
  float s = 0.f;
  for (int i = 0; i < len; i++) s += w[i] * w[i];
  float inv = 1.f / (sqrtf(s) + 1e-8f);
  for (int i = 0; i < len; i++) w[i] *= inv;
}

/* ---- CDNA5 WMMA fragment swizzles (16x16x32 f16, contiguous k = 16*half + e) ---------- */
__device__ __forceinline__ int a_swz(int m, int k, int ksteps) {
  int mt = m >> 4, ks = k >> 5, kk = k & 31;
  int lane = ((kk >> 4) << 4) | (m & 15);
  int e = kk & 15;
  return ((mt * ksteps + ks) * 32 + lane) * 16 + e;
}
__device__ __forceinline__ int b_swz(int k, int n) {  /* 8 n-tiles fixed */
  int ks = k >> 5, kk = k & 31;
  int lane = ((kk >> 4) << 4) | (n & 15);
  int e = kk & 15;
  return ((ks * 8 + (n >> 4)) * 32 + lane) * 16 + e;
}

/* ---- pack normalized weights into the WMMA A-swizzled f16 image (once) ---- */
__global__ void k_pack(const float* __restrict__ p, _Float16* __restrict__ wpk) {
  int idx = blockIdx.x * blockDim.x + threadIdx.x;  /* BATCH*WPK_SZ, r = source order */
  int b = idx / WPK_SZ, r = idx % WPK_SZ;
  int src, dst;
  if (r < 8192)       { int i = r;         int m = i >> 6, k = i & 63;
                        src = OFF_W1 + i; dst = WPK_W1 + a_swz(m, k, 2); }
  else if (r < 24576) { int i = r - 8192;  int m = i >> 7, k = i & 127;
                        src = OFF_W2 + i; dst = WPK_W2 + a_swz(m, k, 4); }
  else if (r < 32768) { int i = r - 24576; int m = i >> 7, k = i & 127;
                        src = OFF_W3 + i; dst = WPK_W3 + a_swz(m, k, 4); }
  else                { int i = r - 32768; int m = i >> 6, k = i & 63;
                        src = OFF_WS + i; dst = WPK_WS + a_swz(m, k, 2); }
  wpk[(size_t)b * WPK_SZ + dst] = (_Float16)p[(size_t)b * FPAR + src];
}

/* --------------- broadcast seed into out_embs slice 0 ---------------------- */
__global__ void k_seed(const float* __restrict__ seed, float* __restrict__ dst) {
  size_t idx = (size_t)blockIdx.x * blockDim.x + threadIdx.x;
  dst[idx] = seed[idx & (size_t)(NF * HW - 1)];
}

/* --------------- conv1x1 64->3 (sob_pre) ----------------------------------- */
__global__ void k_pre(const float* __restrict__ state, const float* __restrict__ w,
                      const float* __restrict__ bias, float* __restrict__ g3) {
  int idx = blockIdx.x * blockDim.x + threadIdx.x;  /* B*HW */
  int b = idx >> 14, pp = idx & (HW - 1);
  const float* xs = state + (size_t)b * NF * HW + pp;
  float a0 = bias[0], a1 = bias[1], a2 = bias[2];
#pragma unroll 8
  for (int i = 0; i < NF; i++) {
    float v = xs[(size_t)i * HW];
    a0 += w[i] * v; a1 += w[NF + i] * v; a2 += w[2 * NF + i] * v;
  }
  g3[((size_t)b * 3 + 0) * HW + pp] = a0;
  g3[((size_t)b * 3 + 1) * HW + pp] = a1;
  g3[((size_t)b * 3 + 2) * HW + pp] = a2;
}

/* ---- SinSobel horizontal pass: t=0 => sin filter (for gx), t=1 => cos ----- */
__global__ void k_sobh(const float* __restrict__ g3, float* __restrict__ tmp) {
  __shared__ float taps[33];
  int idx = blockIdx.x * blockDim.x + threadIdx.x;
  int pix = idx & (HW - 1);
  int rest = idx >> 14;
  int c = rest % 3; rest /= 3;
  int t = rest & 1; rest >>= 1;
  int j = rest % NSCALE;
  int b = rest / NSCALE;
  int k = d_ks[j], pad = k >> 1;
  if ((int)threadIdx.x < k) {
    float r = -1.f + 2.f * (float)threadIdx.x / (float)(k - 1);
    taps[threadIdx.x] = (t == 0) ? sinf(r * PI_2) : cosf(r * PI_2);
  }
  __syncthreads();
  int y = pix >> 7, x = pix & (SS - 1);
  const float* src = g3 + ((size_t)b * 3 + c) * HW + (size_t)y * SS;
  float acc = 0.f;
  for (int d = 0; d < k; d++) {
    int xx = x + d - pad;
    if (xx >= 0 && xx < SS) acc += taps[d] * src[xx];
  }
  tmp[((size_t)b * 30 + j * 6 + t * 3 + c) * HW + pix] = acc;
}

/* ---- SinSobel vertical pass + identity channels --------------------------- */
__global__ void k_sobv(const float* __restrict__ tmp, const float* __restrict__ g3,
                       float* __restrict__ g33) {
  __shared__ float taps[33];
  int idx = blockIdx.x * blockDim.x + threadIdx.x;
  int pix = idx & (HW - 1);
  int rest = idx >> 14;
  int c33 = rest % 33;
  int b = rest / 33;
  if (c33 >= 30) {  /* block-uniform branch: pix is fastest dim */
    g33[((size_t)b * 33 + c33) * HW + pix] = g3[((size_t)b * 3 + (c33 - 30)) * HW + pix];
    return;
  }
  int j = c33 / 6, rem = c33 % 6, t = rem / 3, ch = rem % 3;
  int k = d_ks[j], pad = k >> 1;
  if ((int)threadIdx.x < k) {
    float r = -1.f + 2.f * (float)threadIdx.x / (float)(k - 1);
    taps[threadIdx.x] = (t == 0) ? cosf(r * PI_2) : sinf(r * PI_2);  /* gx: cos vert, gy: sin vert */
  }
  __syncthreads();
  int y = pix >> 7, x = pix & (SS - 1);
  const float* src = tmp + ((size_t)b * 30 + j * 6 + t * 3 + ch) * HW;
  float acc = 0.f;
  for (int d = 0; d < k; d++) {
    int yy = y + d - pad;
    if (yy >= 0 && yy < SS) acc += taps[d] * src[(size_t)yy * SS + x];
  }
  g33[((size_t)b * 33 + c33) * HW + pix] = acc;
}

/* --------------- conv1x1 33->64 (sob_post) --------------------------------- */
__global__ void k_post(const float* __restrict__ g33, const float* __restrict__ w,
                       const float* __restrict__ bias, float* __restrict__ g64) {
  __shared__ float sw[NF * 33];
  __shared__ float sb[NF];
  for (int i = threadIdx.x; i < NF * 33; i += blockDim.x) sw[i] = w[i];
  if (threadIdx.x < NF) sb[threadIdx.x] = bias[threadIdx.x];
  __syncthreads();
  int idx = blockIdx.x * blockDim.x + threadIdx.x;
  int b = idx >> 14, pix = idx & (HW - 1);
  float gv[33];
#pragma unroll
  for (int i = 0; i < 33; i++) gv[i] = g33[((size_t)b * 33 + i) * HW + pix];
  for (int co = 0; co < NF; co++) {
    float acc = sb[co];
#pragma unroll
    for (int i = 0; i < 33; i++) acc += sw[co * 33 + i] * gv[i];
    g64[((size_t)b * NF + co) * HW + pix] = acc;
  }
}

/* ------- gate (state*g) + per-(b,c) instance-norm statistics --------------- */
__global__ void k_gatestat(const float* __restrict__ state, const float* __restrict__ g64,
                           float* __restrict__ xg, float* __restrict__ stats) {
  __shared__ float r1[256], r2[256];
  int bc = blockIdx.x;  /* b*NF + c */
  const float* st = state + (size_t)bc * HW;
  const float* gg = g64 + (size_t)bc * HW;
  float* xo = xg + (size_t)bc * HW;
  float s1 = 0.f, s2 = 0.f;
  for (int p = threadIdx.x; p < HW; p += blockDim.x) {
    float v = st[p] * gg[p];
    xo[p] = v;
    s1 += v; s2 += v * v;
  }
  r1[threadIdx.x] = s1; r2[threadIdx.x] = s2;
  __syncthreads();
  for (int o = 128; o > 0; o >>= 1) {
    if ((int)threadIdx.x < o) { r1[threadIdx.x] += r1[threadIdx.x + o]; r2[threadIdx.x] += r2[threadIdx.x + o]; }
    __syncthreads();
  }
  if (threadIdx.x == 0) {
    float mean = r1[0] * (1.f / HW);
    float var = r2[0] * (1.f / HW) - mean * mean;
    stats[2 * bc]     = mean;
    stats[2 * bc + 1] = rsqrtf(var + 1e-5f);
  }
}

/* ---- fused per-sample hypernet residual block, LDS-resident, WMMA --------- */
/* grid: BATCH * (HW/128) blocks of 256 threads (8 waves). Each block: one    */
/* 128-pixel tile: h1=relu(W1 x+b1); h2=relu(W2 h1+b2); y=W3 h2+b3+Ws x+bs;  */
/* next_state = prev_state + 0.1*y.  Weights arrive pre-swizzled f16 and are */
/* bulk-copied global->LDS (async-to-LDS path when available).               */
__global__ void __launch_bounds__(256)
k_dyna(const _Float16* __restrict__ wpk, const float* __restrict__ p,
       const float* __restrict__ xg, const float* __restrict__ stats,
       const float* __restrict__ prev, float* __restrict__ next) {
  extern __shared__ v16h smem_v[];
  _Float16* sm  = (_Float16*)smem_v;
  _Float16* sA1 = sm + WPK_W1;     /* W1: 128x64  ->  8192 f16 */
  _Float16* sA2 = sm + WPK_W2;     /* W2: 128x128 -> 16384 f16 */
  _Float16* sA3 = sm + WPK_W3;     /* W3: 64x128  ->  8192 f16 */
  _Float16* sAs = sm + WPK_WS;     /* Ws: 64x64   ->  4096 f16 */
  _Float16* sX  = sm + WPK_SZ;     /* x tile 64x128   ->  8192 */
  _Float16* sH1 = sX + 8192;       /* h1 tile 128x128 -> 16384 */
  _Float16* sH2 = sH1 + 16384;     /* h2 tile 128x128 -> 16384 */

  int b = blockIdx.x >> 7;
  int tile = blockIdx.x & 127;
  int base = tile * 128;
  int tid = threadIdx.x;
  const float* pb = p + (size_t)b * FPAR;

  /* ---- bulk copy 72KB of pre-swizzled f16 weights into LDS ---- */
#if USE_ASYNC_COPY
  {
    typedef __attribute__((address_space(1))) v4i* gbl_v4i;
    typedef __attribute__((address_space(3))) v4i* lds_v4i;
    gbl_v4i gsrc = (gbl_v4i)(wpk + (size_t)b * WPK_SZ);
    lds_v4i ldst = (lds_v4i)sm;
    for (int c = tid; c < WPK_SZ / 8; c += 256)   /* 4608 x 16B chunks */
      __builtin_amdgcn_global_load_async_to_lds_b128(gsrc + c, ldst + c, 0, 0);
  }
#else
  {
    const float4* gsrc = (const float4*)(wpk + (size_t)b * WPK_SZ);
    float4* ldst = (float4*)sm;
    for (int c = tid; c < WPK_SZ / 8; c += 256) ldst[c] = gsrc[c];
  }
#endif

  /* ---- x tile: fused instance-norm, f16, B-swizzled, 32B vector stores ---- */
  for (int i = tid; i < 4 * 128; i += 256) {   /* K=64 -> 4 groups of 16 */
    int n = i & 127, g = i >> 7;
    v16h xv;
#pragma unroll
    for (int q = 0; q < 16; q++) {
      int k = g * 16 + q;
      float mean = stats[2 * (b * NF + k)], rstd = stats[2 * (b * NF + k) + 1];
      xv[q] = (_Float16)((xg[((size_t)b * NF + k) * HW + base + n] - mean) * rstd);
    }
    int ks = g >> 1, khalf = g & 1;
    int lane_ = (khalf << 4) | (n & 15);
    int slot = (ks * 8 + (n >> 4)) * 32 + lane_;
    *(v16h*)(&sX[slot * 16]) = xv;
  }

#if USE_ASYNC_COPY
# if defined(__has_builtin) && __has_builtin(__builtin_amdgcn_s_wait_asynccnt)
  __builtin_amdgcn_s_wait_asynccnt(0);
# else
  asm volatile("s_wait_asynccnt 0" ::: "memory");
# endif
#endif
  __syncthreads();

  const v16h* vA1 = (const v16h*)sA1;
  const v16h* vA2 = (const v16h*)sA2;
  const v16h* vA3 = (const v16h*)sA3;
  const v16h* vAs = (const v16h*)sAs;
  const v16h* vX  = (const v16h*)sX;
  const v16h* vH1 = (const v16h*)sH1;
  const v16h* vH2 = (const v16h*)sH2;

  int wv = tid >> 5, lane = tid & 31;
  int lh = lane >> 4, ln = lane & 15;
  int m0 = wv * 16 + 8 * lh;                 /* 8 consecutive D rows per lane */
  v8f zero = {0.f, 0.f, 0.f, 0.f, 0.f, 0.f, 0.f, 0.f};

  /* GEMM1: wave wv owns m rows [16wv,16wv+16), all 8 n-tiles, K=64 */
  {
    v8f acc[8];
#pragma unroll
    for (int nt = 0; nt < 8; nt++) acc[nt] = zero;
#pragma unroll
    for (int ks = 0; ks < 2; ks++) {
      v16h a = vA1[(wv * 2 + ks) * 32 + lane];
#pragma unroll
      for (int nt = 0; nt < 8; nt++) {
        v16h bb = vX[(ks * 8 + nt) * 32 + lane];
        acc[nt] = __builtin_amdgcn_wmma_f32_16x16x32_f16(false, a, false, bb, (short)0, acc[nt], false, false);
      }
    }
    float bias[8];
#pragma unroll
    for (int r = 0; r < 8; r++) bias[r] = pb[OFF_B1 + m0 + r];
#pragma unroll
    for (int nt = 0; nt < 8; nt++) {
      v8h hv;
#pragma unroll
      for (int r = 0; r < 8; r++) hv[r] = (_Float16)fmaxf(acc[nt][r] + bias[r], 0.f);
      *(v8h*)(&sH1[b_swz(m0, nt * 16 + ln)]) = hv;   /* 16B contiguous */
    }
  }
  __syncthreads();

  /* GEMM2: K=128 */
  {
    v8f acc[8];
#pragma unroll
    for (int nt = 0; nt < 8; nt++) acc[nt] = zero;
#pragma unroll
    for (int ks = 0; ks < 4; ks++) {
      v16h a = vA2[(wv * 4 + ks) * 32 + lane];
#pragma unroll
      for (int nt = 0; nt < 8; nt++) {
        v16h bb = vH1[(ks * 8 + nt) * 32 + lane];
        acc[nt] = __builtin_amdgcn_wmma_f32_16x16x32_f16(false, a, false, bb, (short)0, acc[nt], false, false);
      }
    }
    float bias[8];
#pragma unroll
    for (int r = 0; r < 8; r++) bias[r] = pb[OFF_B2 + m0 + r];
#pragma unroll
    for (int nt = 0; nt < 8; nt++) {
      v8h hv;
#pragma unroll
      for (int r = 0; r < 8; r++) hv[r] = (_Float16)fmaxf(acc[nt][r] + bias[r], 0.f);
      *(v8h*)(&sH2[b_swz(m0, nt * 16 + ln)]) = hv;
    }
  }
  __syncthreads();

  /* GEMM3 (W3 @ h2, K=128) + shortcut (Ws @ x, K=64); M=64 -> 4 m-tiles.     */
  /* wave wv: m-tile = wv>>1, n-tiles (wv&1)*4 .. +3                          */
  {
    int mt = wv >> 1, nh = wv & 1;
    int c0 = mt * 16 + 8 * lh;
    v8f acc[4];
#pragma unroll
    for (int i = 0; i < 4; i++) acc[i] = zero;
#pragma unroll
    for (int ks = 0; ks < 4; ks++) {
      v16h a = vA3[(mt * 4 + ks) * 32 + lane];
#pragma unroll
      for (int i = 0; i < 4; i++) {
        v16h bb = vH2[(ks * 8 + nh * 4 + i) * 32 + lane];
        acc[i] = __builtin_amdgcn_wmma_f32_16x16x32_f16(false, a, false, bb, (short)0, acc[i], false, false);
      }
    }
#pragma unroll
    for (int ks = 0; ks < 2; ks++) {
      v16h a = vAs[(mt * 2 + ks) * 32 + lane];
#pragma unroll
      for (int i = 0; i < 4; i++) {
        v16h bb = vX[(ks * 8 + nh * 4 + i) * 32 + lane];
        acc[i] = __builtin_amdgcn_wmma_f32_16x16x32_f16(false, a, false, bb, (short)0, acc[i], false, false);
      }
    }
    float bias[8];
#pragma unroll
    for (int r = 0; r < 8; r++) bias[r] = pb[OFF_B3 + c0 + r] + pb[OFF_BS + c0 + r];
#pragma unroll
    for (int i = 0; i < 4; i++) {
#pragma unroll
      for (int r = 0; r < 8; r++) {
        int c = c0 + r;
        int n = (nh * 4 + i) * 16 + ln;
        float v = acc[i][r] + bias[r];
        size_t o = ((size_t)b * NF + c) * HW + base + n;
        next[o] = prev[o] + 0.1f * v;
      }
    }
  }
}

/* --------------- final residual block + head ------------------------------- */
__global__ void k_res1(const float* __restrict__ state, const float* __restrict__ w,
                       const float* __restrict__ bias, float* __restrict__ y1) {
  __shared__ float sw[NF];
  int idx = blockIdx.x * blockDim.x + threadIdx.x;
  int pix = idx & (HW - 1);
  int co = (idx >> 14) & (NF - 1);
  int b = idx >> 20;
  if (threadIdx.x < NF) sw[threadIdx.x] = w[co * NF + threadIdx.x];
  __syncthreads();
  const float* xs = state + (size_t)b * NF * HW + pix;
  float acc = bias[co];
#pragma unroll 8
  for (int i = 0; i < NF; i++) acc += sw[i] * xs[(size_t)i * HW];
  y1[((size_t)b * NF + co) * HW + pix] = fmaxf(acc, 0.f);
}

__global__ void k_res2(const float* __restrict__ y1, const float* __restrict__ state,
                       const float* __restrict__ w2, const float* __restrict__ b2,
                       const float* __restrict__ wsc, const float* __restrict__ bsc,
                       float* __restrict__ y) {
  __shared__ float sw2[NF], swc[NF];
  int idx = blockIdx.x * blockDim.x + threadIdx.x;
  int pix = idx & (HW - 1);
  int co = (idx >> 14) & (NF - 1);
  int b = idx >> 20;
  if (threadIdx.x < NF) { sw2[threadIdx.x] = w2[co * NF + threadIdx.x]; swc[threadIdx.x] = wsc[co * NF + threadIdx.x]; }
  __syncthreads();
  const float* ys = y1 + (size_t)b * NF * HW + pix;
  const float* xs = state + (size_t)b * NF * HW + pix;
  float acc = b2[co] + bsc[co];
#pragma unroll 8
  for (int i = 0; i < NF; i++) acc += sw2[i] * ys[(size_t)i * HW] + swc[i] * xs[(size_t)i * HW];
  y[((size_t)b * NF + co) * HW + pix] = acc;
}

__global__ void k_out(const float* __restrict__ y, const float* __restrict__ w,
                      const float* __restrict__ bias, float* __restrict__ img,
                      float* __restrict__ raw) {
  int idx = blockIdx.x * blockDim.x + threadIdx.x;
  int b = idx >> 14, pix = idx & (HW - 1);
  const float* ys = y + (size_t)b * NF * HW + pix;
  float a0 = bias[0], a1 = bias[1], a2 = bias[2];
#pragma unroll 8
  for (int i = 0; i < NF; i++) {
    float v = ys[(size_t)i * HW];
    a0 += w[i] * v; a1 += w[NF + i] * v; a2 += w[2 * NF + i] * v;
  }
  size_t o = ((size_t)b * 3) * HW + pix;
  raw[o] = a0;          raw[o + HW] = a1;          raw[o + 2 * HW] = a2;
  img[o] = fminf(fmaxf(a0, -1.f), 1.f);
  img[o + HW] = fminf(fmaxf(a1, -1.f), 1.f);
  img[o + 2 * HW] = fminf(fmaxf(a2, -1.f), 1.f);
}

extern "C" void kernel_launch(void* const* d_in, const int* in_sizes, int n_in,
                              void* d_out, int out_size, void* d_ws, size_t ws_size,
                              hipStream_t stream) {
  (void)in_sizes; (void)n_in; (void)out_size; (void)ws_size;
  const float* lat        = (const float*)d_in[0];
  const float* seed       = (const float*)d_in[1];
  const float* sob_pre_w  = (const float*)d_in[2];
  const float* sob_pre_b  = (const float*)d_in[3];
  const float* sob_post_w = (const float*)d_in[4];
  const float* sob_post_b = (const float*)d_in[5];
  const float* dyna_w     = (const float*)d_in[6];
  const float* dyna_b     = (const float*)d_in[7];
  const float* res_w1     = (const float*)d_in[8];
  const float* res_b1     = (const float*)d_in[9];
  const float* res_w2     = (const float*)d_in[10];
  const float* res_b2     = (const float*)d_in[11];
  const float* res_wsc    = (const float*)d_in[12];
  const float* res_bsc    = (const float*)d_in[13];
  const float* out_w      = (const float*)d_in[14];
  const float* out_b      = (const float*)d_in[15];

  float* out      = (float*)d_out;
  float* out_img  = out;                                   /* B*3*HW */
  float* out_embs = out + (size_t)BATCH * CHO * HW;        /* (NC+1)*EMB */
  float* out_raw  = out_embs + (size_t)(NC + 1) * EMB;     /* B*3*HW */

  float* ws = (float*)d_ws;
  size_t o = 0;
  float* p     = ws + o; o += (size_t)BATCH * FPAR;        /* 595,968  */
  float* stats = ws + o; o += (size_t)2 * BATCH * NF;      /* 2,048    */
  float* g3    = ws + o; o += (size_t)BATCH * 3 * HW;      /* 786,432  */
  float* tmp   = ws + o; o += (size_t)BATCH * 30 * HW;     /* 7,864,320 */
  float* g33   = ws + o; o += (size_t)BATCH * 33 * HW;     /* 8,650,752 */
  float* g64   = ws + o; o += (size_t)BATCH * NF * HW;     /* 16,777,216 */
  float* xg    = ws + o; o += (size_t)BATCH * NF * HW;     /* 16,777,216 */
  _Float16* wpk = (_Float16*)(ws + o); o += (size_t)BATCH * WPK_SZ / 2;  /* 294,912 */
  float* y1b   = g64;   /* reused after the step loop */
  float* yb    = xg;    /* reused after the step loop */

  /* one-time: hypernet params, weight norm, swizzled f16 pack, seed */
  k_hyper<<<(FPAR + 255) / 256, 256, 0, stream>>>(lat, dyna_w, dyna_b, p);
  k_wnorm<<<(BATCH * 384 + 255) / 256, 256, 0, stream>>>(p);
  k_pack<<<BATCH * WPK_SZ / 256, 256, 0, stream>>>(p, wpk);
  k_seed<<<(int)(EMB / 256), 256, 0, stream>>>(seed, out_embs);

  size_t dynaLds = (size_t)77824 * sizeof(_Float16);  /* 152 KB */
  for (int t = 0; t < NC; t++) {
    const float* st = out_embs + (size_t)t * EMB;
    float* nx = out_embs + (size_t)(t + 1) * EMB;
    k_pre <<<BATCH * HW / 256, 256, 0, stream>>>(st, sob_pre_w, sob_pre_b, g3);
    k_sobh<<<BATCH * NSCALE * 2 * 3 * HW / 256, 256, 0, stream>>>(g3, tmp);
    k_sobv<<<BATCH * 33 * HW / 256, 256, 0, stream>>>(tmp, g3, g33);
    k_post<<<BATCH * HW / 256, 256, 0, stream>>>(g33, sob_post_w, sob_post_b, g64);
    k_gatestat<<<BATCH * NF, 256, 0, stream>>>(st, g64, xg, stats);
    k_dyna<<<BATCH * (HW / 128), 256, dynaLds, stream>>>(wpk, p, xg, stats, st, nx);
  }

  const float* fin = out_embs + (size_t)NC * EMB;
  k_res1<<<BATCH * NF * HW / 256, 256, 0, stream>>>(fin, res_w1, res_b1, y1b);
  k_res2<<<BATCH * NF * HW / 256, 256, 0, stream>>>(y1b, fin, res_w2, res_b2, res_wsc, res_bsc, yb);
  k_out <<<BATCH * HW / 256, 256, 0, stream>>>(yb, out_w, out_b, out_img, out_raw);
}
